// FlexMACE_1778116461342
// MI455X (gfx1250) — compile-verified, matrix operand order checked by reading, not run on hardware
//
#include <hip/hip_runtime.h>
#include <math.h>

// ---------------------------------------------------------------------------
// MACE-style GNN forward for MI455X (gfx1250, wave32, WMMA).
// Edge MLP (~200 GFLOP) on v_wmma_f32_16x16x32_f16 with fragment-swizzled
// weights (2x global_load_b128 per B frag) and fragment-layout LDS staging
// (2x ds_load_b128 per A frag). 32 edges per block, 2 waves split N-range,
// each B fragment reused for two M-tiles.
// ---------------------------------------------------------------------------

typedef _Float16 v16h __attribute__((ext_vector_type(16)));
typedef _Float16 v4h  __attribute__((ext_vector_type(4)));
typedef float    v8f  __attribute__((ext_vector_type(8)));

#define F 64
#define RB 32
#define DIN 161          // 2F + RB + 1
#define K1 192           // DIN padded to multiple of 32
#define HDIM 256
#define OUT3 192         // 3F
#define NSPEC 5
#define NLAYER 2
#define KT1 6            // 192/32 k-tiles
#define KT2 8            // 256/32 k-tiles
#define NT12 16          // 256/16 n-tiles (GEMM1, GEMM2)
#define NT3 12           // 192/16 n-tiles (GEMM3)

__device__ __forceinline__ float silu_f(float x) {
    return x / (1.0f + __expf(-x));
}

__device__ __forceinline__ v8f wmma32(v16h a, v16h b, v8f c) {
    return __builtin_amdgcn_wmma_f32_16x16x32_f16(false, a, false, b,
                                                  (short)0, c, false, false);
}

// Fragment-position helpers (16-bit A layout, ISA 7.12.2):
// element k_local (0..31) of a fragment lives in lane-half hp=(kl>>3)&1 at
// half-index i16 = (kl&7) | ((kl&16)>>1) inside that lane's 16 contiguous halfs.
__device__ __forceinline__ int frag_i16(int kl) {
    return (kl & 7) | ((kl & 16) >> 1);
}
__device__ __forceinline__ int frag_hp(int kl) {
    return (kl >> 3) & 1;
}
// A-matrix fragment element (tile m-index tm, k-tile kt, row m, k_local kl)
__device__ __forceinline__ int afrag_off(int tm, int KT, int kt, int m, int kl) {
    return ((tm * KT + kt) * 32 + (m | (frag_hp(kl) << 4))) * 16 + frag_i16(kl);
}
// B-matrix fragment element (k-tile kt, n-tile nt, n_local nl, k_local kl)
__device__ __forceinline__ int bfrag_off(int kt, int nt, int ntiles, int nl, int kl) {
    return ((kt * ntiles + nt) * 32 + (nl | (frag_hp(kl) << 4))) * 16 + frag_i16(kl);
}

// ------------------------- tiny helpers ------------------------------------

__global__ void k_zero(float* __restrict__ p, long long n) {
    long long i  = (long long)blockIdx.x * blockDim.x + threadIdx.x;
    long long st = (long long)gridDim.x * blockDim.x;
    for (; i < n; i += st) p[i] = 0.0f;
}

__global__ void k_embed(const float* __restrict__ shared_f,
                        const int*   __restrict__ spec,
                        const float* __restrict__ embed,
                        const float* __restrict__ W_sf,
                        const float* __restrict__ b_sf,
                        float* __restrict__ s, int N) {
    int idx = blockIdx.x * blockDim.x + threadIdx.x;
    int n = idx >> 6, f = idx & 63;
    if (n >= N) return;
    float acc = embed[spec[n] * F + f] + b_sf[f];
    const float* sr = shared_f + (size_t)n * F;
    #pragma unroll 8
    for (int k = 0; k < F; ++k) acc += sr[k] * W_sf[k * F + f];
    s[(size_t)n * F + f] = acc;
}

__global__ void k_geom(const float* __restrict__ pos,
                       const int* __restrict__ snd, const int* __restrict__ rcv,
                       float* __restrict__ sh, float* __restrict__ len_,
                       float* __restrict__ radial, int E) {
    int e = blockIdx.x * blockDim.x + threadIdx.x;
    if (e >= E) return;
    int a = snd[e], b = rcv[e];
    float vx = pos[b * 3 + 0] - pos[a * 3 + 0];
    float vy = pos[b * 3 + 1] - pos[a * 3 + 1];
    float vz = pos[b * 3 + 2] - pos[a * 3 + 2];
    const float EPS = 1e-6f;
    float l = sqrtf(vx * vx + vy * vy + vz * vz + EPS * EPS);
    float inv = 1.0f / fmaxf(l, EPS);
    const float SQ3 = 1.7320508075688772f;
    sh[e * 3 + 0] = SQ3 * vx * inv;
    sh[e * 3 + 1] = SQ3 * vy * inv;
    sh[e * 3 + 2] = SQ3 * vz * inv;
    len_[e] = l;
    #pragma unroll 8
    for (int j = 0; j < RB; ++j) {
        float mu = (4.0f / 31.0f) * (float)j;
        float d = (l - mu) * 8.0f;
        radial[(size_t)e * RB + j] = __expf(-0.5f * d * d);
    }
}

// Convert one layer's MLP weights to f16 *in B-fragment layout*.
// W1: 161x256 zero-padded to 192x256; W2: 256x256; W3: 256x192.
__global__ void k_cvt_w(const float* __restrict__ W1, const float* __restrict__ W2,
                        const float* __restrict__ W3,
                        _Float16* __restrict__ W1f, _Float16* __restrict__ W2f,
                        _Float16* __restrict__ W3f) {
    const int N1 = K1 * HDIM, N2 = HDIM * HDIM, N3 = HDIM * OUT3;
    int i = blockIdx.x * blockDim.x + threadIdx.x;
    if (i < N1) {
        int k = i / HDIM, n = i - k * HDIM;
        float v = (k < DIN) ? W1[k * HDIM + n] : 0.0f;
        W1f[bfrag_off(k >> 5, n >> 4, NT12, n & 15, k & 31)] = (_Float16)v;
    } else if (i < N1 + N2) {
        int j = i - N1;
        int k = j / HDIM, n = j - k * HDIM;
        W2f[bfrag_off(k >> 5, n >> 4, NT12, n & 15, k & 31)] = (_Float16)W2[j];
    } else if (i < N1 + N2 + N3) {
        int j = i - N1 - N2;
        int k = j / OUT3, n = j - k * OUT3;
        W3f[bfrag_off(k >> 5, n >> 4, NT3, n & 15, k & 31)] = (_Float16)W3[j];
    }
}

__global__ void k_node_up(const float* __restrict__ s, const float* __restrict__ v,
                          const float* __restrict__ upW, const float* __restrict__ upb,
                          const float* __restrict__ upWv,
                          float* __restrict__ s1, float* __restrict__ v1, int N) {
    int idx = blockIdx.x * blockDim.x + threadIdx.x;
    int n = idx >> 6, f = idx & 63;
    if (n >= N) return;
    const float* sr = s + (size_t)n * F;
    float acc = upb[f];
    #pragma unroll 8
    for (int k = 0; k < F; ++k) acc += sr[k] * upW[k * F + f];
    s1[(size_t)n * F + f] = acc;
    const float* vr = v + (size_t)n * F * 3;
    #pragma unroll
    for (int c = 0; c < 3; ++c) {
        float t = 0.0f;
        #pragma unroll 8
        for (int k = 0; k < F; ++k) t += vr[k * 3 + c] * upWv[k * F + f];
        v1[((size_t)n * F + f) * 3 + c] = t;
    }
}

// ------------------------- fused WMMA edge kernel ---------------------------
// Block = 64 threads (2 waves), 32 edges (two 16-row M-tiles).
// Stage h (32x192) in fragment layout -> GEMM1+SiLU -> GEMM2+SiLU -> GEMM3
// -> messages + atomic scatter. Waves split the N-tile range; each B fragment
// (one contiguous 32B chunk per lane) is reused for both M-tiles.
__global__ __launch_bounds__(64) void k_edge_wmma(
    const float* __restrict__ s1, const float* __restrict__ v1,
    const int* __restrict__ snd, const int* __restrict__ rcv,
    const float* __restrict__ sh, const float* __restrict__ len_,
    const float* __restrict__ radial,
    const _Float16* __restrict__ W1f, const float* __restrict__ b1,
    const _Float16* __restrict__ W2f, const float* __restrict__ b2,
    const _Float16* __restrict__ W3f, const float* __restrict__ b3,
    float* __restrict__ a_s, float* __restrict__ a_v, int E) {
    __shared__ _Float16 fb0[2 * KT2 * 512];   // 16 KB
    __shared__ _Float16 fb1[2 * KT2 * 512];   // 16 KB
    const int tid = threadIdx.x;
    const int wv = tid >> 5, lane = tid & 31;
    const int hh = lane >> 4;
    const int e0 = blockIdx.x * 32;

    // ---- stage h (32 x 192) into fb0, A-fragment layout (KT=6) ----
    for (int q = tid; q < 32 * 48; q += 64) {
        int row = q / 48, c4 = (q - row * 48) * 4;
        int e = e0 + row;
        float x0 = 0.f, x1 = 0.f, x2 = 0.f, x3 = 0.f;
        if (e < E) {
            if (c4 < F) {
                const float4 t = *(const float4*)(s1 + (size_t)snd[e] * F + c4);
                x0 = t.x; x1 = t.y; x2 = t.z; x3 = t.w;
            } else if (c4 < 2 * F) {
                const float4 t = *(const float4*)(s1 + (size_t)rcv[e] * F + (c4 - F));
                x0 = t.x; x1 = t.y; x2 = t.z; x3 = t.w;
            } else if (c4 < 2 * F + RB) {
                const float4 t = *(const float4*)(radial + (size_t)e * RB + (c4 - 2 * F));
                x0 = t.x; x1 = t.y; x2 = t.z; x3 = t.w;
            } else if (c4 == DIN - 1) {
                x0 = len_[e];
            }
        }
        int tm = row >> 4, m = row & 15;
        int kt = c4 >> 5, kl = c4 & 31;
        v4h h4 = { (_Float16)x0, (_Float16)x1, (_Float16)x2, (_Float16)x3 };
        *(v4h*)(fb0 + afrag_off(tm, KT1, kt, m, kl)) = h4;
    }
    __syncthreads();

    // ---- GEMM1 (16x192 @ 192x256) + SiLU -> fb1 (KT=8 layout) ----
    #pragma unroll 1
    for (int nt = wv * 8; nt < wv * 8 + 8; ++nt) {
        v8f acc0 = {}, acc1 = {};
        #pragma unroll
        for (int ks = 0; ks < KT1; ++ks) {
            v16h b  = *(const v16h*)(W1f + ((size_t)(ks * NT12 + nt) * 32 + lane) * 16);
            v16h a0 = *(const v16h*)(fb0 + ((0 * KT1 + ks) * 32 + lane) * 16);
            v16h a1 = *(const v16h*)(fb0 + ((1 * KT1 + ks) * 32 + lane) * 16);
            acc0 = wmma32(a0, b, acc0);
            acc1 = wmma32(a1, b, acc1);
        }
        int n = nt * 16 + (lane & 15);
        float bias = b1[n];
        int kt = n >> 5, kl = n & 31;
        #pragma unroll
        for (int r = 0; r < 8; ++r) {
            int ml = r + 8 * hh;
            fb1[afrag_off(0, KT2, kt, ml, kl)] = (_Float16)silu_f(acc0[r] + bias);
            fb1[afrag_off(1, KT2, kt, ml, kl)] = (_Float16)silu_f(acc1[r] + bias);
        }
    }
    __syncthreads();

    // ---- GEMM2 (16x256 @ 256x256) + SiLU -> fb0 (KT=8 layout) ----
    #pragma unroll 1
    for (int nt = wv * 8; nt < wv * 8 + 8; ++nt) {
        v8f acc0 = {}, acc1 = {};
        #pragma unroll
        for (int ks = 0; ks < KT2; ++ks) {
            v16h b  = *(const v16h*)(W2f + ((size_t)(ks * NT12 + nt) * 32 + lane) * 16);
            v16h a0 = *(const v16h*)(fb1 + ((0 * KT2 + ks) * 32 + lane) * 16);
            v16h a1 = *(const v16h*)(fb1 + ((1 * KT2 + ks) * 32 + lane) * 16);
            acc0 = wmma32(a0, b, acc0);
            acc1 = wmma32(a1, b, acc1);
        }
        int n = nt * 16 + (lane & 15);
        float bias = b2[n];
        int kt = n >> 5, kl = n & 31;
        #pragma unroll
        for (int r = 0; r < 8; ++r) {
            int ml = r + 8 * hh;
            fb0[afrag_off(0, KT2, kt, ml, kl)] = (_Float16)silu_f(acc0[r] + bias);
            fb0[afrag_off(1, KT2, kt, ml, kl)] = (_Float16)silu_f(acc1[r] + bias);
        }
    }
    __syncthreads();

    // ---- GEMM3 (16x256 @ 256x192) -> mix; messages + atomic scatter ----
    #pragma unroll 1
    for (int nt = wv * 6; nt < wv * 6 + 6; ++nt) {
        v8f acc[2];
        acc[0] = (v8f){}; acc[1] = (v8f){};
        #pragma unroll
        for (int ks = 0; ks < KT2; ++ks) {
            v16h b  = *(const v16h*)(W3f + ((size_t)(ks * NT3 + nt) * 32 + lane) * 16);
            v16h a0 = *(const v16h*)(fb0 + ((0 * KT2 + ks) * 32 + lane) * 16);
            v16h a1 = *(const v16h*)(fb0 + ((1 * KT2 + ks) * 32 + lane) * 16);
            acc[0] = wmma32(a0, b, acc[0]);
            acc[1] = wmma32(a1, b, acc[1]);
        }
        int col = nt * 16 + (lane & 15);
        float bias = b3[col];
        int grp = nt >> 2;              // 0: m_s, 1: m_v1, 2: m_v2
        int f = col - grp * F;
        #pragma unroll 1
        for (int tm = 0; tm < 2; ++tm) {
            #pragma unroll 1
            for (int r = 0; r < 8; ++r) {
                int e = e0 + tm * 16 + r + 8 * hh;
                if (e >= E) continue;
                float mix = acc[tm][r] + bias;
                int se = snd[e], re = rcv[e];
                float shx = sh[e * 3 + 0], shy = sh[e * 3 + 1], shz = sh[e * 3 + 2];
                if (grp == 0) {
                    const float* vs = v1 + ((size_t)se * F + f) * 3;
                    float ms = (vs[0] * shx + vs[1] * shy + vs[2] * shz)
                               * 0.5773502691896258f * mix;     // /sqrt(3)
                    atomicAdd(a_s + (size_t)re * F + f, ms);
                } else if (grp == 1) {
                    float t = s1[(size_t)se * F + f] * mix;
                    float* dst = a_v + ((size_t)re * 2 * F + f) * 3;
                    atomicAdd(dst + 0, t * shx);
                    atomicAdd(dst + 1, t * shy);
                    atomicAdd(dst + 2, t * shz);
                } else {
                    const float* vs = v1 + ((size_t)se * F + f) * 3;
                    float k2 = 1.224744871391589f * mix;        // sqrt(3/2)
                    float cx = (vs[1] * shz - vs[2] * shy) * k2;
                    float cy = (vs[2] * shx - vs[0] * shz) * k2;
                    float cz = (vs[0] * shy - vs[1] * shx) * k2;
                    float* dst = a_v + ((size_t)re * 2 * F + F + f) * 3;
                    atomicAdd(dst + 0, cx);
                    atomicAdd(dst + 1, cy);
                    atomicAdd(dst + 2, cz);
                }
            }
        }
    }
}

// ------------------- per-node post (down / poly / readout) ------------------
__global__ __launch_bounds__(64) void k_node_post(
    const float* __restrict__ a_s, const float* __restrict__ a_v,
    const int* __restrict__ spec,
    const float* __restrict__ downW, const float* __restrict__ downb,
    const float* __restrict__ downWv,
    const float* __restrict__ scWs, const float* __restrict__ scWv,
    const float* __restrict__ slW, const float* __restrict__ slb,
    const float* __restrict__ slWv,
    const float* __restrict__ roW, const float* __restrict__ rob,
    const float* __restrict__ roWv,
    const float* __restrict__ outW, const float* __restrict__ outb,
    const float* __restrict__ outWv,
    float* __restrict__ s, float* __restrict__ v,
    float* __restrict__ out, int N) {
    __shared__ float as_[F], av_[2 * F * 3];
    __shared__ float s2_[F], v2_[F * 3];
    __shared__ float cs_[F], cv_[F * 3];
    __shared__ float s3_[F], v3_[F * 3];
    __shared__ float r_[2 * F];
    __shared__ float rs_[F], rv_[F * 3];
    __shared__ float red_[F * 4];

    int n = blockIdx.x;
    if (n >= N) return;
    int f = threadIdx.x;

    as_[f] = a_s[(size_t)n * F + f] * (1.0f / 16.0f);      // / AVG_NN
    for (int i = f; i < 2 * F * 3; i += F)
        av_[i] = a_v[(size_t)n * 2 * F * 3 + i] * (1.0f / 16.0f);
    __syncthreads();

    {
        float acc = downb[f];
        #pragma unroll 8
        for (int k = 0; k < F; ++k) acc += as_[k] * downW[k * F + f];
        s2_[f] = acc;
        #pragma unroll
        for (int c = 0; c < 3; ++c) {
            float t = 0.0f;
            #pragma unroll 8
            for (int k = 0; k < 2 * F; ++k) t += av_[k * 3 + c] * downWv[k * F + f];
            v2_[f * 3 + c] = t;
        }
    }
    __syncthreads();

    {
        int sp = spec[n];
        const float* ws = scWs + ((size_t)sp * F + f) * 5;
        const float* wv = scWv + ((size_t)sp * F + f) * 4;
        float x = s2_[f];
        float vx = v2_[f * 3], vy = v2_[f * 3 + 1], vz = v2_[f * 3 + 2];
        float vv = vx * vx + vy * vy + vz * vz;
        cs_[f] = ws[0] * x + ws[1] * x * x + ws[2] * vv + ws[3] * x * x * x
               + ws[4] * x * vv;
        float g = wv[0] + wv[1] * x + wv[2] * x * x + wv[3] * vv;
        cv_[f * 3 + 0] = g * vx;
        cv_[f * 3 + 1] = g * vy;
        cv_[f * 3 + 2] = g * vz;
    }
    __syncthreads();

    {
        float acc = slb[f];
        #pragma unroll 8
        for (int k = 0; k < F; ++k) acc += cs_[k] * slW[k * F + f];
        s3_[f] = acc;
        s[(size_t)n * F + f] = acc;
        #pragma unroll
        for (int c = 0; c < 3; ++c) {
            float t = 0.0f;
            #pragma unroll 8
            for (int k = 0; k < F; ++k) t += cv_[k * 3 + c] * slWv[k * F + f];
            v3_[f * 3 + c] = t;
            v[((size_t)n * F + f) * 3 + c] = t;
        }
    }
    __syncthreads();

    for (int j = f; j < 2 * F; j += F) {
        float t = rob[j];
        #pragma unroll 8
        for (int k = 0; k < F; ++k) t += s3_[k] * roW[k * 2 * F + j];
        r_[j] = t;
    }
    __syncthreads();

    {
        rs_[f] = silu_f(r_[f]);
        float gate = silu_f(r_[F + f]);
        #pragma unroll
        for (int c = 0; c < 3; ++c) {
            float t = 0.0f;
            #pragma unroll 8
            for (int k = 0; k < F; ++k) t += v3_[k * 3 + c] * roWv[k * F + f];
            rv_[f * 3 + c] = t * gate;
        }
    }
    __syncthreads();

    red_[f * 4 + 0] = rs_[f] * outW[f];
    red_[f * 4 + 1] = rv_[f * 3 + 0] * outWv[f];
    red_[f * 4 + 2] = rv_[f * 3 + 1] * outWv[f];
    red_[f * 4 + 3] = rv_[f * 3 + 2] * outWv[f];
    __syncthreads();
    if (f < 4) {
        float t = (f == 0) ? outb[0] : 0.0f;
        for (int k = 0; k < F; ++k) t += red_[k * 4 + f];
        out[(size_t)n * 4 + f] += t;
    }
}

// ---------------------------------------------------------------------------

extern "C" void kernel_launch(void* const* d_in, const int* in_sizes, int n_in,
                              void* d_out, int out_size, void* d_ws, size_t ws_size,
                              hipStream_t stream) {
    (void)n_in; (void)out_size; (void)ws_size;
    const float* positions = (const float*)d_in[0];
    const int*   spec      = (const int*)d_in[1];
    const float* shared_f  = (const float*)d_in[2];
    const int*   senders   = (const int*)d_in[3];
    const int*   receivers = (const int*)d_in[4];
    const float* embed     = (const float*)d_in[5];
    const float* W_sf      = (const float*)d_in[6];
    const float* b_sf      = (const float*)d_in[7];
    const float* up_Ws     = (const float*)d_in[8];
    const float* up_bs     = (const float*)d_in[9];
    const float* up_Wv     = (const float*)d_in[10];
    const float* mlp_W1    = (const float*)d_in[11];
    const float* mlp_b1    = (const float*)d_in[12];
    const float* mlp_W2    = (const float*)d_in[13];
    const float* mlp_b2    = (const float*)d_in[14];
    const float* mlp_W3    = (const float*)d_in[15];
    const float* mlp_b3    = (const float*)d_in[16];
    const float* down_Ws   = (const float*)d_in[17];
    const float* down_bs   = (const float*)d_in[18];
    const float* down_Wv   = (const float*)d_in[19];
    const float* sc_Ws     = (const float*)d_in[20];
    const float* sc_Wv     = (const float*)d_in[21];
    const float* sc_lin_Ws = (const float*)d_in[22];
    const float* sc_lin_bs = (const float*)d_in[23];
    const float* sc_lin_Wv = (const float*)d_in[24];
    const float* ro_Ws     = (const float*)d_in[25];
    const float* ro_bs     = (const float*)d_in[26];
    const float* ro_Wv     = (const float*)d_in[27];
    const float* out_Ws    = (const float*)d_in[28];
    const float* out_bs    = (const float*)d_in[29];
    const float* out_Wv    = (const float*)d_in[30];

    const int N = in_sizes[0] / 3;
    const int E = in_sizes[3];
    float* out = (float*)d_out;

    char* base = (char*)d_ws;
    size_t off = 0;
    auto carve = [&](size_t bytes) -> void* {
        void* r = base + off;
        off += (bytes + 255) & ~(size_t)255;
        return r;
    };
    float* s      = (float*)carve((size_t)N * F * 4);
    float* v      = (float*)carve((size_t)N * F * 3 * 4);
    float* s1     = (float*)carve((size_t)N * F * 4);
    float* v1     = (float*)carve((size_t)N * F * 3 * 4);
    float* a_s    = (float*)carve((size_t)N * F * 4);
    float* a_v    = (float*)carve((size_t)N * 2 * F * 3 * 4);
    float* shbuf  = (float*)carve((size_t)E * 3 * 4);
    float* lenbuf = (float*)carve((size_t)E * 4);
    float* radbuf = (float*)carve((size_t)E * RB * 4);
    _Float16* W1f = (_Float16*)carve((size_t)K1 * HDIM * 2);
    _Float16* W2f = (_Float16*)carve((size_t)HDIM * HDIM * 2);
    _Float16* W3f = (_Float16*)carve((size_t)HDIM * OUT3 * 2);

    k_zero<<<1024, 256, 0, stream>>>(out, (long long)N * 4);
    k_zero<<<1024, 256, 0, stream>>>(v, (long long)N * F * 3);

    k_embed<<<(N * F + 255) / 256, 256, 0, stream>>>(shared_f, spec, embed, W_sf,
                                                     b_sf, s, N);
    k_geom<<<(E + 255) / 256, 256, 0, stream>>>(positions, senders, receivers,
                                                shbuf, lenbuf, radbuf, E);

    const int eblocks = (E + 31) / 32;
    const int cvt_total = K1 * HDIM + HDIM * HDIM + HDIM * OUT3;

    for (int l = 0; l < NLAYER; ++l) {
        k_node_up<<<(N * F + 255) / 256, 256, 0, stream>>>(
            s, v, up_Ws + (size_t)l * F * F, up_bs + (size_t)l * F,
            up_Wv + (size_t)l * F * F, s1, v1, N);

        k_zero<<<1024, 256, 0, stream>>>(a_s, (long long)N * F);
        k_zero<<<1024, 256, 0, stream>>>(a_v, (long long)N * 2 * F * 3);

        k_cvt_w<<<(cvt_total + 255) / 256, 256, 0, stream>>>(
            mlp_W1 + (size_t)l * DIN * HDIM, mlp_W2 + (size_t)l * HDIM * HDIM,
            mlp_W3 + (size_t)l * HDIM * OUT3, W1f, W2f, W3f);

        k_edge_wmma<<<eblocks, 64, 0, stream>>>(
            s1, v1, senders, receivers, shbuf, lenbuf, radbuf,
            W1f, mlp_b1 + (size_t)l * HDIM,
            W2f, mlp_b2 + (size_t)l * HDIM,
            W3f, mlp_b3 + (size_t)l * OUT3,
            a_s, a_v, E);

        k_node_post<<<N, 64, 0, stream>>>(
            a_s, a_v, spec,
            down_Ws + (size_t)l * F * F, down_bs + (size_t)l * F,
            down_Wv + (size_t)l * 2 * F * F,
            sc_Ws + (size_t)l * NSPEC * F * 5, sc_Wv + (size_t)l * NSPEC * F * 4,
            sc_lin_Ws + (size_t)l * F * F, sc_lin_bs + (size_t)l * F,
            sc_lin_Wv + (size_t)l * F * F,
            ro_Ws + (size_t)l * F * 2 * F, ro_bs + (size_t)l * 2 * F,
            ro_Wv + (size_t)l * F * F,
            out_Ws + (size_t)l * F, out_bs + (size_t)l,
            out_Wv + (size_t)l * F,
            s, v, out, N);
    }
}
